// GCN_10282151707324
// MI455X (gfx1250) — compile-verified, matrix operand order checked by reading, not run on hardware
//
#include <hip/hip_runtime.h>
#include <math.h>

// ---------------- problem constants ----------------
#define HID   256
#define MIDC  128
#define OUTC  16
#define NGRAPH 512
#define EPSLN 1e-5f

typedef _Float16 v8h  __attribute__((ext_vector_type(8)));
typedef _Float16 v16h __attribute__((ext_vector_type(16)));
typedef float    v8f  __attribute__((ext_vector_type(8)));

// ---------------- degree / norm ----------------
__global__ void k_deg(const int* __restrict__ dst, float* __restrict__ deg, int E) {
    int e = blockIdx.x * blockDim.x + threadIdx.x;
    if (e < E) atomicAdd(&deg[dst[e]], 1.0f);
}

__global__ void k_dinv(float* __restrict__ deg, int N) {
    int i = blockIdx.x * blockDim.x + threadIdx.x;
    if (i < N) deg[i] = rsqrtf(deg[i] + 1.0f);   // +1 = self loop; always > 0
}

// ---------------- conv1: h = pos @ W1  (K=2) ----------------
__global__ void k_conv1(const float* __restrict__ pos, const float* __restrict__ w,
                        float* __restrict__ h, long long n) {
    long long t = blockIdx.x * (long long)blockDim.x + threadIdx.x;
    if (t >= n) return;
    int i = (int)(t >> 8), c = (int)(t & 255);
    h[t] = fmaf(pos[2 * i], w[c], pos[2 * i + 1] * w[HID + c]);
}

// ---------------- aggregation: init with bias + self-loop ----------------
__global__ void k_init_agg(const float* __restrict__ h, const float* __restrict__ dinv,
                           const float* __restrict__ bias, float* __restrict__ out,
                           long long n) {
    long long t = blockIdx.x * (long long)blockDim.x + threadIdx.x;
    if (t >= n) return;
    int i = (int)(t >> 8), c = (int)(t & 255);
    float di = dinv[i];
    out[t] = fmaf(h[t], di * di, bias[c]);
}

// ---------------- aggregation: edges, 4 channels per thread ----------------
__global__ void k_agg_edges(const float* __restrict__ h, const float* __restrict__ dinv,
                            const int* __restrict__ src, const int* __restrict__ dst,
                            float* __restrict__ out, long long nwork) {
    long long t = blockIdx.x * (long long)blockDim.x + threadIdx.x;
    if (t >= nwork) return;
    int e  = (int)(t >> 6);          // HID/4 = 64 chunks per edge
    int c4 = (int)(t & 63) << 2;
    int s = src[e], d = dst[e];
    float nrm = dinv[s] * dinv[d];
    const float4 hv = *(const float4*)(h + (long long)s * HID + c4);
    float* o = out + (long long)d * HID + c4;
    atomicAdd(o + 0, hv.x * nrm);
    atomicAdd(o + 1, hv.y * nrm);
    atomicAdd(o + 2, hv.z * nrm);
    atomicAdd(o + 3, hv.w * nrm);
}

// ---------------- graph layernorm: partial sums ----------------
__global__ void k_reduce(const float* __restrict__ x, long long n, float* __restrict__ s) {
    __shared__ float s1[256], s2[256];
    float a = 0.f, b = 0.f;
    for (long long i = blockIdx.x * (long long)blockDim.x + threadIdx.x; i < n;
         i += (long long)gridDim.x * blockDim.x) {
        float v = x[i];
        a += v; b = fmaf(v, v, b);
    }
    s1[threadIdx.x] = a; s2[threadIdx.x] = b;
    __syncthreads();
    for (int off = 128; off > 0; off >>= 1) {
        if (threadIdx.x < off) {
            s1[threadIdx.x] += s1[threadIdx.x + off];
            s2[threadIdx.x] += s2[threadIdx.x + off];
        }
        __syncthreads();
    }
    if (threadIdx.x == 0) { atomicAdd(&s[0], s1[0]); atomicAdd(&s[1], s2[0]); }
}

__global__ void k_ln_finalize(float* __restrict__ s, float n) {
    float mean = s[0] / n;
    float var  = fmaxf(s[1] / n - mean * mean, 0.0f);
    s[2] = mean;
    s[3] = 1.0f / (sqrtf(var) + EPSLN);
}

// ---------------- fused normalize + affine + PReLU (+optional f16 cast) ----------------
__global__ void k_ln_prelu(float* __restrict__ x, const float* __restrict__ w,
                           const float* __restrict__ b, const float* __restrict__ alpha,
                           const float* __restrict__ stats, int cmask, long long n,
                           _Float16* __restrict__ xh) {
    long long t = blockIdx.x * (long long)blockDim.x + threadIdx.x;
    if (t >= n) return;
    int c = (int)(t & cmask);
    float v = (x[t] - stats[2]) * stats[3];
    v = fmaf(v, w[c], b[c]);
    v = (v >= 0.f) ? v : alpha[0] * v;
    x[t] = v;
    if (xh) xh[t] = (_Float16)v;
}

// ---------------- transpose + f16 convert of W2 [K,N] -> BT [N,K] ----------------
__global__ void k_wT_f16(const float* __restrict__ w, _Float16* __restrict__ bt) {
    int t = blockIdx.x * blockDim.x + threadIdx.x;   // 256*256
    if (t >= HID * HID) return;
    int n = t >> 8, k = t & 255;
    bt[n * HID + k] = (_Float16)w[k * HID + n];
}

// ============================================================================
// WMMA GEMM: C[M,256] = A[M,256]f16 * BT[256,256]f16^T
//  - block = 256 threads (8 waves); blockIdx = (mblk of 8 row-tiles) x (4 col-groups)
//  - B staged into LDS (32 KB: 64 rows x 256 halfs) via global_load_async_to_lds_b128
//    (ASYNCcnt path), drained with s_wait_asynccnt, then read via ds_load_b128.
//  - each wave computes a 16x64 tile: one A fragment feeds 4 WMMA accumulators.
// ============================================================================
__global__ __launch_bounds__(256) void k_gemm_wmma(
        const _Float16* __restrict__ A, const _Float16* __restrict__ BT,
        float* __restrict__ C, int M) {
    __shared__ _Float16 Bs[64 * HID];          // 32 KB
    const int tid  = threadIdx.x;
    const int ngrp = blockIdx.x & 3;           // which 64-column group
    const int mblk = blockIdx.x >> 2;          // which group of 8 row-tiles

    // ---- async DMA: stage 64 BT rows (full K) into LDS ----
    {
        const _Float16* gsrc = BT + (long long)ngrp * 64 * HID;
#pragma unroll
        for (int i = 0; i < 8; ++i) {
            int idx = tid + i * 256;                       // 2048 x 16B = 32 KB
            const _Float16* g = gsrc + idx * 8;
            unsigned lo = (unsigned)(unsigned long long)(Bs + idx * 8); // LDS offset (low 32b of generic ptr)
            asm volatile("global_load_async_to_lds_b128 %0, %1, off"
                         :: "v"(lo), "v"(g) : "memory");
        }
        asm volatile("s_wait_asynccnt 0x0" ::: "memory");
    }
    __syncthreads();

    const int wave = tid >> 5;
    const int lane = tid & 31;
    const int r    = lane & 15;                // row (A) / col (B) within 16-tile
    const int hh   = lane >> 4;                // selects K half-blocks
    const int mt   = mblk * 8 + wave;
    int ma = mt * 16 + r; if (ma >= M) ma = M - 1;         // clamp tail tiles
    const _Float16* arow = A  + (long long)ma * HID + hh * 8;
    const _Float16* brow = Bs + r * HID + hh * 8;          // sub-tile 0 base (LDS)

    v8f acc[4];
#pragma unroll
    for (int s = 0; s < 4; ++s) acc[s] = (v8f){0.f,0.f,0.f,0.f,0.f,0.f,0.f,0.f};

#pragma unroll
    for (int kb = 0; kb < 8; ++kb) {                       // K = 256 = 8 x 32
        v8h a0 = *(const v8h*)(arow + kb * 32);
        v8h a1 = *(const v8h*)(arow + kb * 32 + 16);
        v16h av = __builtin_shufflevector(a0, a1, 0,1,2,3,4,5,6,7,8,9,10,11,12,13,14,15);
#pragma unroll
        for (int s = 0; s < 4; ++s) {
            const _Float16* bp = brow + s * 16 * HID + kb * 32;
            v8h b0 = *(const v8h*)(bp);
            v8h b1 = *(const v8h*)(bp + 16);
            v16h bv = __builtin_shufflevector(b0, b1, 0,1,2,3,4,5,6,7,8,9,10,11,12,13,14,15);
            acc[s] = __builtin_amdgcn_wmma_f32_16x16x32_f16(false, av, false, bv,
                                                            (short)0, acc[s], false, false);
        }
    }

    const int mb = mt * 16 + 8 * hh;
#pragma unroll
    for (int s = 0; s < 4; ++s) {
        int n = ngrp * 64 + s * 16 + r;
#pragma unroll
        for (int v = 0; v < 8; ++v) {
            int m = mb + v;
            if (m < M) C[(long long)m * HID + n] = acc[s][v];
        }
    }
}

// ---------------- global mean pool (sums + counts) ----------------
__global__ void k_pool(const float* __restrict__ x, const int* __restrict__ batch,
                       float* __restrict__ pool, float* __restrict__ cnt, long long nwork) {
    long long t = blockIdx.x * (long long)blockDim.x + threadIdx.x;
    if (t >= nwork) return;
    int i  = (int)(t >> 6);
    int c4 = (int)(t & 63) << 2;
    int g = batch[i];
    const float4 v = *(const float4*)(x + (long long)i * HID + c4);
    float* p = pool + (long long)g * HID + c4;
    atomicAdd(p + 0, v.x); atomicAdd(p + 1, v.y);
    atomicAdd(p + 2, v.z); atomicAdd(p + 3, v.w);
    if (c4 == 0) atomicAdd(&cnt[g], 1.0f);
}

// ---------------- lin1: mid[512,128] = (pool/cnt) @ W[256,128] + b ----------------
__global__ void k_lin1(const float* __restrict__ pool, const float* __restrict__ cnt,
                       const float* __restrict__ w, const float* __restrict__ b,
                       float* __restrict__ mid) {
    int t = blockIdx.x * blockDim.x + threadIdx.x;   // 512*128
    if (t >= NGRAPH * MIDC) return;
    int g = t >> 7, j = t & 127;
    float inv = 1.0f / fmaxf(cnt[g], 1.0f);
    float acc = b[j];
    const float* pg = pool + g * HID;
    for (int k = 0; k < HID; ++k) acc = fmaf(pg[k] * inv, w[k * MIDC + j], acc);
    mid[t] = acc;
}

// ---------------- lin2: out[512,16] = mid @ W[128,16] + b ----------------
__global__ void k_lin2(const float* __restrict__ mid, const float* __restrict__ w,
                       const float* __restrict__ b, float* __restrict__ out) {
    int t = blockIdx.x * blockDim.x + threadIdx.x;   // 512*16
    if (t >= NGRAPH * OUTC) return;
    int g = t >> 4, o = t & 15;
    float acc = b[o];
    const float* mg = mid + g * MIDC;
    for (int j = 0; j < MIDC; ++j) acc = fmaf(mg[j], w[j * OUTC + o], acc);
    out[t] = acc;
}

// ---------------- host ----------------
static inline size_t alignup(size_t x) { return (x + 255) & ~(size_t)255; }

extern "C" void kernel_launch(void* const* d_in, const int* in_sizes, int n_in,
                              void* d_out, int out_size, void* d_ws, size_t ws_size,
                              hipStream_t stream) {
    const float* pos   = (const float*)d_in[0];
    const int*   eidx  = (const int*)d_in[1];
    const int*   batch = (const int*)d_in[2];
    const float* w1    = (const float*)d_in[3];
    const float* b1    = (const float*)d_in[4];
    const float* ln1w  = (const float*)d_in[5];
    const float* ln1b  = (const float*)d_in[6];
    const float* a1    = (const float*)d_in[7];
    const float* w2    = (const float*)d_in[8];
    const float* b2    = (const float*)d_in[9];
    const float* ln2w  = (const float*)d_in[10];
    const float* ln2b  = (const float*)d_in[11];
    const float* a2    = (const float*)d_in[12];
    const float* wl1   = (const float*)d_in[13];
    const float* bl1   = (const float*)d_in[14];
    const float* lnmw  = (const float*)d_in[15];
    const float* lnmb  = (const float*)d_in[16];
    const float* am    = (const float*)d_in[17];
    const float* wl2   = (const float*)d_in[18];
    const float* bl2   = (const float*)d_in[19];

    const int N = in_sizes[0] / 2;          // nodes
    const int E = in_sizes[1] / 2;          // edges
    const int* src = eidx;
    const int* dst = eidx + E;

    // workspace carve-up
    char* p = (char*)d_ws;
    size_t off = 0;
    float*    scr  = (float*)(p + off);                off += alignup(48 * sizeof(float));
    float*    dinv = (float*)(p + off);                off += alignup((size_t)N * 4);
    float*    H    = (float*)(p + off);                off += alignup((size_t)N * HID * 4);
    float*    X    = (float*)(p + off);                off += alignup((size_t)N * HID * 4);
    _Float16* Xh   = (_Float16*)(p + off);             off += alignup((size_t)N * HID * 2);
    _Float16* BT   = (_Float16*)(p + off);             off += alignup((size_t)HID * HID * 2);
    float*    pool = (float*)(p + off);                off += alignup((size_t)NGRAPH * HID * 4);
    float*    cnt  = (float*)(p + off);                off += alignup((size_t)NGRAPH * 4);
    float*    mid  = (float*)(p + off);                off += alignup((size_t)NGRAPH * MIDC * 4);
    (void)ws_size;

    const long long nNC   = (long long)N * HID;
    const long long nEdgW = (long long)E * (HID / 4);
    const int B256 = 256;
    auto blk = [](long long n, int b) { return (unsigned)((n + b - 1) / b); };

    // ---- degree / dinv ----
    hipMemsetAsync(dinv, 0, (size_t)N * 4, stream);
    k_deg <<<blk(E, B256), B256, 0, stream>>>(dst, dinv, E);
    k_dinv<<<blk(N, B256), B256, 0, stream>>>(dinv, N);

    // ---- conv1 ----
    k_conv1   <<<blk(nNC, B256), B256, 0, stream>>>(pos, w1, H, nNC);
    k_init_agg<<<blk(nNC, B256), B256, 0, stream>>>(H, dinv, b1, X, nNC);
    k_agg_edges<<<blk(nEdgW, B256), B256, 0, stream>>>(H, dinv, src, dst, X, nEdgW);

    // ---- LN1 + PReLU (+ f16 cast for WMMA GEMM) ----
    hipMemsetAsync(scr, 0, 16, stream);
    k_reduce     <<<4096, B256, 0, stream>>>(X, nNC, scr);
    k_ln_finalize<<<1, 1, 0, stream>>>(scr, (float)nNC);
    k_ln_prelu   <<<blk(nNC, B256), B256, 0, stream>>>(X, ln1w, ln1b, a1, scr, 255, nNC, Xh);

    // ---- conv2: WMMA GEMM H = Xh @ W2 ----
    k_wT_f16<<<blk(HID * HID, B256), B256, 0, stream>>>(w2, BT);
    {
        int mtiles  = (N + 15) / 16;
        int mblocks = (mtiles + 7) / 8;
        k_gemm_wmma<<<(unsigned)(mblocks * 4), 256, 0, stream>>>(Xh, BT, H, N);
    }
    k_init_agg <<<blk(nNC, B256), B256, 0, stream>>>(H, dinv, b2, X, nNC);
    k_agg_edges<<<blk(nEdgW, B256), B256, 0, stream>>>(H, dinv, src, dst, X, nEdgW);

    // ---- LN2 + PReLU ----
    hipMemsetAsync(scr + 16, 0, 16, stream);
    k_reduce     <<<4096, B256, 0, stream>>>(X, nNC, scr + 16);
    k_ln_finalize<<<1, 1, 0, stream>>>(scr + 16, (float)nNC);
    k_ln_prelu   <<<blk(nNC, B256), B256, 0, stream>>>(X, ln2w, ln2b, a2, scr + 16, 255,
                                                       nNC, (_Float16*)nullptr);

    // ---- global mean pool ----
    hipMemsetAsync(pool, 0, (size_t)NGRAPH * HID * 4, stream);
    hipMemsetAsync(cnt, 0, (size_t)NGRAPH * 4, stream);
    {
        long long nw = (long long)N * (HID / 4);
        k_pool<<<blk(nw, B256), B256, 0, stream>>>(X, batch, pool, cnt, nw);
    }

    // ---- lin1 -> LNm + PReLU -> lin2 ----
    k_lin1<<<blk(NGRAPH * MIDC, B256), B256, 0, stream>>>(pool, cnt, wl1, bl1, mid);

    const long long nM = (long long)NGRAPH * MIDC;
    hipMemsetAsync(scr + 32, 0, 16, stream);
    k_reduce     <<<256, B256, 0, stream>>>(mid, nM, scr + 32);
    k_ln_finalize<<<1, 1, 0, stream>>>(scr + 32, (float)nM);
    k_ln_prelu   <<<blk(nM, B256), B256, 0, stream>>>(mid, lnmw, lnmb, am, scr + 32, 127,
                                                      nM, (_Float16*)nullptr);

    k_lin2<<<blk(NGRAPH * OUTC, B256), B256, 0, stream>>>(mid, wl2, bl2, (float*)d_out);
}
